// PointHeadPseudoLab_25451976196113
// MI455X (gfx1250) — matrix operand order repaired and checked by reading, not compiled, not run on hardware
//
#include <hip/hip_runtime.h>

typedef __attribute__((ext_vector_type(16))) __bf16 v16bf;
typedef __attribute__((ext_vector_type(8)))  float  v8f;

#define LQ_TOT 19950
#define NB 2
#define DMODEL 256

#define BM 128
#define BN 64
#define BK 32
#define LDS_P (BK + 8)

// round-half-up f32 -> bf16 (cheap: one add, take high half)
__device__ __forceinline__ unsigned short f2bf(float f) {
  unsigned int u = __builtin_bit_cast(unsigned int, f) + 0x8000u;
  return (unsigned short)(u >> 16);
}
// pack two f32 into two bf16 in one DWORD: v_add x2 + v_perm_b32
__device__ __forceinline__ unsigned int pack2bf(float a, float b) {
  unsigned int ua = __builtin_bit_cast(unsigned int, a) + 0x8000u;
  unsigned int ub = __builtin_bit_cast(unsigned int, b) + 0x8000u;
  return __builtin_amdgcn_perm(ub, ua, 0x07060302u);  // {ub.hi16, ua.hi16}
}
__device__ __forceinline__ __bf16 us2bf(unsigned short u) {
  return __builtin_bit_cast(__bf16, u);
}
// LDS byte offset of a __shared__ pointer (LDS aperture: addr[31:0])
__device__ __forceinline__ unsigned lds_off(const void* p) {
  return (unsigned)(unsigned long)p;
}
// async 16B copy global -> LDS, tracked by ASYNCcnt
__device__ __forceinline__ void async_cp16(unsigned lds_b, const void* gptr) {
  asm volatile("global_load_async_to_lds_b128 %0, %1, off"
               :: "v"(lds_b), "v"((unsigned long)gptr) : "memory");
}
__device__ __forceinline__ void wait_async0() {
  asm volatile("s_wait_asynccnt 0x0" ::: "memory");
}

// ---------------------------------------------------------------------------
// f32 -> bf16 bulk converter (weights; n multiple of 4)
// ---------------------------------------------------------------------------
__global__ __launch_bounds__(256) void cvt_bf16_kernel(
    const float* __restrict__ in, unsigned short* __restrict__ out, long n4)
{
  long i = (long)blockIdx.x * 256 + threadIdx.x;
  if (i >= n4) return;
  float4 v = ((const float4*)in)[i];
  uint2 u;
  u.x = pack2bf(v.x, v.y);
  u.y = pack2bf(v.z, v.w);
  ((uint2*)out)[i] = u;
}

// ---------------------------------------------------------------------------
// GEMM: C[m,n] = sum_k A[m,k] * B[n,k] + bias[n]   (optional ReLU)
// B is bf16 [Nn][K] (pre-converted weights) -> async DMA tiles to LDS.
// ABF=1: A is bf16, k-contiguous (token GEMMs) -> async DMA tiles to LDS.
// ABF=0: A is f32, k-strided (conv projections) -> coalesced clamped loads + cvt.
// Output: f32, or bf16 when obf16 (FFN intermediate).
// Block = 256 thr = 8 waves; tile 128x64; double-buffered LDS, 1 barrier/tile.
// ---------------------------------------------------------------------------
template <int ABF>
__global__ __launch_bounds__(256) void gemm_wmma(
    const void* __restrict__ Av, const unsigned short* __restrict__ Bh,
    const float* __restrict__ bias, void* __restrict__ Cv,
    int M, int Nn, int K,
    long a_m_stride, long a_k_stride, long a_batch_stride,
    long c_batch_stride, int relu, int obf16)
{
  __shared__ alignas(16) unsigned short sA[2][BM * LDS_P];
  __shared__ alignas(16) unsigned short sB[2][BN * LDS_P];

  const int tid  = threadIdx.x;
  const int lane = tid & 31;
  const int wv   = tid >> 5;
  const int m0   = blockIdx.y * BM;
  const int n0   = blockIdx.x * BN;
  const float*          Af = (const float*)Av + (ABF ? 0 : (long)blockIdx.z * a_batch_stride);
  const unsigned short* Ah = (const unsigned short*)Av + (ABF ? (long)blockIdx.z * a_batch_stride : 0);

  const v8f zero = {0.f,0.f,0.f,0.f,0.f,0.f,0.f,0.f};
  v8f acc[4] = {zero, zero, zero, zero};

  float ras[16];    // ABF=0 staging only

  // B tile (64x32 bf16 = 4 KB): one async 16B chunk per thread
  auto copyB_async = [&](int k0, int buf) {
    int r   = tid >> 2;                   // col 0..63
    int ko  = (tid & 3) * 8;              // k oct
    int col = n0 + r; if (col > Nn - 1) col = Nn - 1;
    async_cp16(lds_off(&sB[buf][r * LDS_P + ko]),
               Bh + (long)col * K + (k0 + ko));
  };
  // A tile (128x32 bf16 = 8 KB): two async 16B chunks per thread
  auto copyA_async = [&](int k0, int buf) {
    #pragma unroll
    for (int i = 0; i < 2; ++i) {
      int idx = tid + 256 * i;            // 0..511 octs
      int r   = idx >> 2;                 // row 0..127
      int ko  = (idx & 3) * 8;
      int m   = m0 + r; if (m > M - 1) m = M - 1;
      async_cp16(lds_off(&sA[buf][r * LDS_P + ko]),
                 Ah + (long)m * a_m_stride + (k0 + ko));
    }
  };
  // f32 strided A path (conv projections): coalesced clamped loads + cvt
  auto loadA_f32 = [&](int k0) {
    #pragma unroll
    for (int i = 0; i < 16; ++i) {
      int idx = tid + 256 * i;            // 0..4095
      int r   = idx & 127;                // consecutive tid -> consecutive m
      int k   = idx >> 7;
      int m   = m0 + r; if (m > M - 1) m = M - 1;
      ras[i] = Af[(long)m * a_m_stride + (long)(k0 + k) * a_k_stride];
    }
  };
  auto storeA_f32 = [&](int buf) {
    #pragma unroll
    for (int i = 0; i < 16; ++i) {
      int idx = tid + 256 * i;
      int r   = idx & 127;
      int k   = idx >> 7;
      sA[buf][r * LDS_P + k] = f2bf(ras[i]);
    }
  };

  const int nT = K >> 5;                  // K is a multiple of 32 for all calls
  copyB_async(0, 0);
  if (ABF) copyA_async(0, 0);
  else     { loadA_f32(0); storeA_f32(0); }
  wait_async0();
  __syncthreads();

  for (int t = 0; t < nT; ++t) {
    const int cur = t & 1;
    if (t + 1 < nT) {
      copyB_async((t + 1) * BK, cur ^ 1);
      if (ABF) copyA_async((t + 1) * BK, cur ^ 1);
      else     loadA_f32((t + 1) * BK);
    }
    if (t + 2 < nT) {                     // speculative prefetch 2 tiles ahead
      int r = tid >> 1;
      int m = m0 + (r < BM ? r : BM - 1);
      if (m > M - 1) m = M - 1;
      if (ABF) __builtin_prefetch(Ah + (long)m * a_m_stride + (long)((t + 2) * BK), 0, 3);
      else     __builtin_prefetch(Af + (long)m * a_m_stride + (long)((t + 2) * BK) * a_k_stride, 0, 3);
    }

    // A fragment: 16-bit A 16x32 layout (lanes 0-15: K 0-7/16-23; lanes 16-31: K 8-15/24-31)
    v16bf af;
    {
      const unsigned short* pr = &sA[cur][(wv * 16 + (lane & 15)) * LDS_P];
      const int kb = (lane >> 4) * 8;
      #pragma unroll
      for (int i = 0; i < 8; ++i) af[i] = us2bf(pr[kb + i]);
      #pragma unroll
      for (int i = 0; i < 8; ++i) af[8 + i] = us2bf(pr[16 + kb + i]);
    }
    #pragma unroll
    for (int j = 0; j < 4; ++j) {
      // B fragment: 16-bit B 32x16 layout (lanes 0-15: K 0-7/16-23; lanes 16-31: K 8-15/24-31)
      v16bf bf;
      const unsigned short* pc = &sB[cur][(j * 16 + (lane & 15)) * LDS_P];
      const int kb2 = (lane >> 4) * 8;
      #pragma unroll
      for (int i = 0; i < 8; ++i) { bf[i] = us2bf(pc[kb2 + i]); bf[8 + i] = us2bf(pc[16 + kb2 + i]); }
      acc[j] = __builtin_amdgcn_wmma_f32_16x16x32_bf16(
          false, af, false, bf, (short)0, acc[j], false, false);
    }

    if (t + 1 < nT) {
      if (!ABF) storeA_f32(cur ^ 1);
      wait_async0();                      // async tile for t+1 has landed
    }
    __syncthreads();
  }

  // C/D layout: lane VGPR v -> row = v + 8*(lane>>4), col = lane&15
  float*          Cf = (float*)Cv          + (long)blockIdx.z * c_batch_stride;
  unsigned short* Ch = (unsigned short*)Cv + (long)blockIdx.z * c_batch_stride;
  const int rbase = m0 + wv * 16 + (lane >> 4) * 8;
  #pragma unroll
  for (int j = 0; j < 4; ++j) {
    int col = n0 + j * 16 + (lane & 15);
    if (col < Nn) {
      float bv = bias ? bias[col] : 0.f;
      #pragma unroll
      for (int v = 0; v < 8; ++v) {
        int m = rbase + v;
        if (m < M) {
          float o = acc[j][v] + bv;
          if (relu) o = fmaxf(o, 0.f);
          if (obf16) Ch[(long)m * Nn + col] = f2bf(o);
          else       Cf[(long)m * Nn + col] = o;
        }
      }
    }
  }
}

// ---------------------------------------------------------------------------
// GroupNorm stats over one level: per (n, group of 8 channels) across HW.
// ---------------------------------------------------------------------------
__global__ __launch_bounds__(256) void gn_stats_kernel(
    const float* __restrict__ P, float* __restrict__ stats, int HW, int startRow)
{
  __shared__ float rs[256], rs2[256];
  int g = blockIdx.x, n = blockIdx.y, t = threadIdx.x;
  int c = g * 8 + (t & 7);
  float s = 0.f, s2 = 0.f;
  for (int r = (t >> 3); r < HW; r += 32) {
    float v = P[((long)(n * LQ_TOT + startRow + r)) * DMODEL + c];
    s += v; s2 += v * v;
  }
  rs[t] = s; rs2[t] = s2;
  __syncthreads();
  for (int o = 128; o > 0; o >>= 1) {
    if (t < o) { rs[t] += rs[t + o]; rs2[t] += rs2[t + o]; }
    __syncthreads();
  }
  if (t == 0) {
    float cnt = 8.f * (float)HW;
    float mean = rs[0] / cnt;
    float var  = rs2[0] / cnt - mean * mean;
    stats[(n * 32 + g) * 2 + 0] = mean;
    stats[(n * 32 + g) * 2 + 1] = rsqrtf(var + 1e-5f);
  }
}

// ---------------------------------------------------------------------------
// Apply GN -> src (f32 + bf16); sine positional encoding + level embed -> q (bf16)
// ---------------------------------------------------------------------------
__global__ __launch_bounds__(256) void gn_apply_pos_kernel(
    const float* __restrict__ P, const float* __restrict__ stats,
    const float* __restrict__ gamma, const float* __restrict__ beta,
    const float* __restrict__ lvl_embed, float* __restrict__ src,
    unsigned short* __restrict__ src16, unsigned short* __restrict__ q16,
    int HW, int startRow, int Hh, int Ww)
{
  long idx = (long)blockIdx.x * 256 + threadIdx.x;
  if (idx >= (long)HW * DMODEL) return;
  int n = blockIdx.y;
  int r = (int)(idx >> 8);
  int c = (int)(idx & 255);
  long o = ((long)(n * LQ_TOT + startRow + r)) * DMODEL + c;
  float mean = stats[(n * 32 + (c >> 3)) * 2 + 0];
  float rstd = stats[(n * 32 + (c >> 3)) * 2 + 1];
  float s = (P[o] - mean) * rstd * gamma[c] + beta[c];
  src[o]   = s;
  src16[o] = f2bf(s);
  int iy = r / Ww, ix = r - iy * Ww;
  int cc = c & 127;
  float base = (c < 128) ? ((float)(iy + 1) * 6.28318530718f / ((float)Hh + 1e-6f))
                         : ((float)(ix + 1) * 6.28318530718f / ((float)Ww + 1e-6f));
  float dimt = __expf(9.210340372f * (float)(2 * (cc >> 1)) * (1.0f / 128.0f)); // 10000^(2(cc/2)/128)
  float v = base / dimt;
  float posv = (cc & 1) ? __cosf(v) : __sinf(v);
  q16[o] = f2bf(s + posv + lvl_embed[c]);
}

// ---------------------------------------------------------------------------
// Deformable attention sampling: one wave per (n, query, head); lane = channel.
// Writes accumulator as bf16 (consumed only by the output-projection GEMM).
// ---------------------------------------------------------------------------
__global__ __launch_bounds__(256) void deform_sample_kernel(
    const float* __restrict__ value, const float* __restrict__ offv,
    const float* __restrict__ awl, unsigned short* __restrict__ accb)
{
  const int lvlH[3] = {100, 50, 25};
  const int lvlW[3] = {152, 76, 38};
  const int lvlS[3] = {0, 15200, 19000};
  int wvid = blockIdx.x * 8 + (threadIdx.x >> 5);
  int lane = threadIdx.x & 31;
  if (wvid >= NB * LQ_TOT * 8) return;
  int hd = wvid & 7;
  int t  = wvid >> 3;
  int qi = t % LQ_TOT;
  int n  = t / LQ_TOT;
  long row = (long)n * LQ_TOT + qi;

  int l0, li;
  if (qi < 15200)      { l0 = 0; li = qi; }
  else if (qi < 19000) { l0 = 1; li = qi - 15200; }
  else                 { l0 = 2; li = qi - 19000; }
  float refx = ((float)(li % lvlW[l0]) + 0.5f) / (float)lvlW[l0];
  float refy = ((float)(li / lvlW[l0]) + 0.5f) / (float)lvlH[l0];

  float lw[12];
  float mx = -1e30f;
  #pragma unroll
  for (int j = 0; j < 12; ++j) {
    lw[j] = awl[row * 96 + hd * 12 + j];
    mx = fmaxf(mx, lw[j]);
  }
  float ssum = 0.f;
  #pragma unroll
  for (int j = 0; j < 12; ++j) { lw[j] = __expf(lw[j] - mx); ssum += lw[j]; }
  float inv = 1.f / ssum;

  float accv = 0.f;
  #pragma unroll
  for (int lvl = 0; lvl < 3; ++lvl) {
    int W = lvlW[lvl], H = lvlH[lvl], st = lvlS[lvl];
    #pragma unroll
    for (int pt = 0; pt < 4; ++pt) {
      int j = lvl * 4 + pt;
      float ox = offv[row * 192 + hd * 24 + lvl * 8 + pt * 2 + 0];
      float oy = offv[row * 192 + hd * 24 + lvl * 8 + pt * 2 + 1];
      float x = (refx + ox / (float)W) * (float)W - 0.5f;
      float y = (refy + oy / (float)H) * (float)H - 0.5f;
      float xf = floorf(x), yf = floorf(y);
      float wx = x - xf, wy = y - yf;
      int x0 = (int)xf, y0 = (int)yf;
      float smp = 0.f;
      #pragma unroll
      for (int dy = 0; dy < 2; ++dy) {
        #pragma unroll
        for (int dx = 0; dx < 2; ++dx) {
          int xi = x0 + dx, yi = y0 + dy;
          if (xi >= 0 && xi < W && yi >= 0 && yi < H) {
            float wgt = (dx ? wx : 1.f - wx) * (dy ? wy : 1.f - wy);
            long vrow = (long)n * LQ_TOT + st + yi * W + xi;
            smp += wgt * value[vrow * DMODEL + hd * 32 + lane];
          }
        }
      }
      accv += (lw[j] * inv) * smp;
    }
  }
  accb[row * DMODEL + hd * 32 + lane] = f2bf(accv);
}

// ---------------------------------------------------------------------------
// out = LayerNorm(a + b) * g + be ; optional extra bf16 copy (GEMM input)
// ---------------------------------------------------------------------------
__global__ __launch_bounds__(256) void add_ln_kernel(
    const float* __restrict__ a, const float* __restrict__ b,
    const float* __restrict__ g, const float* __restrict__ be,
    float* __restrict__ out, unsigned short* __restrict__ out16)
{
  __shared__ float red[256];
  long row = blockIdx.x;
  int t = threadIdx.x;
  float v = a[row * DMODEL + t] + b[row * DMODEL + t];
  red[t] = v; __syncthreads();
  for (int o = 128; o > 0; o >>= 1) { if (t < o) red[t] += red[t + o]; __syncthreads(); }
  float mean = red[0] * (1.0f / DMODEL);
  __syncthreads();
  float d = v - mean;
  red[t] = d * d; __syncthreads();
  for (int o = 128; o > 0; o >>= 1) { if (t < o) red[t] += red[t + o]; __syncthreads(); }
  float rstd = rsqrtf(red[0] * (1.0f / DMODEL) + 1e-5f);
  float r = d * rstd * g[t] + be[t];
  out[row * DMODEL + t] = r;
  if (out16) out16[row * DMODEL + t] = f2bf(r);
}

// ---------------------------------------------------------------------------
// Bilinear-sample level-0 memory at point coords -> bf16 rows for final FC.
// ---------------------------------------------------------------------------
__global__ __launch_bounds__(256) void point_sample_kernel(
    const float* __restrict__ x2, const float* __restrict__ pts,
    unsigned short* __restrict__ pf)
{
  long idx = (long)blockIdx.x * 256 + threadIdx.x;
  if (idx >= (long)NB * 1000 * DMODEL) return;
  int c = (int)(idx & 255);
  long pr = idx >> 8;                 // n*1000 + p
  int n = (int)(pr / 1000);
  const int W = 152, H = 100;
  float x = pts[pr * 2 + 0] * (float)W - 0.5f;
  float y = pts[pr * 2 + 1] * (float)H - 0.5f;
  float xf = floorf(x), yf = floorf(y);
  float wx = x - xf, wy = y - yf;
  int x0 = (int)xf, y0 = (int)yf;
  float r = 0.f;
  #pragma unroll
  for (int dy = 0; dy < 2; ++dy)
    #pragma unroll
    for (int dx = 0; dx < 2; ++dx) {
      int xi = x0 + dx, yi = y0 + dy;
      if (xi >= 0 && xi < W && yi >= 0 && yi < H) {
        float wgt = (dx ? wx : 1.f - wx) * (dy ? wy : 1.f - wy);
        r += wgt * x2[((long)n * LQ_TOT + yi * W + xi) * DMODEL + c];
      }
    }
  pf[pr * DMODEL + c] = f2bf(r);
}

// ---------------------------------------------------------------------------
extern "C" void kernel_launch(void* const* d_in, const int* in_sizes, int n_in,
                              void* d_out, int out_size, void* d_ws, size_t ws_size,
                              hipStream_t stream)
{
  const float* feat[3]   = {(const float*)d_in[0], (const float*)d_in[1], (const float*)d_in[2]};
  const float* pcoords   = (const float*)d_in[3];
  const void*  gtcls     = d_in[4];
  const float* proj_w[3] = {(const float*)d_in[5], (const float*)d_in[9],  (const float*)d_in[13]};
  const float* proj_b[3] = {(const float*)d_in[6], (const float*)d_in[10], (const float*)d_in[14]};
  const float* gn_g[3]   = {(const float*)d_in[7], (const float*)d_in[11], (const float*)d_in[15]};
  const float* gn_b[3]   = {(const float*)d_in[8], (const float*)d_in[12], (const float*)d_in[16]};
  const float* level_emb = (const float*)d_in[17];
  const float* w_off  = (const float*)d_in[18]; const float* b_off  = (const float*)d_in[19];
  const float* w_attn = (const float*)d_in[20]; const float* b_attn = (const float*)d_in[21];
  const float* w_val  = (const float*)d_in[22]; const float* b_val  = (const float*)d_in[23];
  const float* w_out  = (const float*)d_in[24]; const float* b_out  = (const float*)d_in[25];
  const float* ln1_g  = (const float*)d_in[26]; const float* ln1_b  = (const float*)d_in[27];
  const float* ln2_g  = (const float*)d_in[28]; const float* ln2_b  = (const float*)d_in[29];
  const float* w_ff1  = (const float*)d_in[30]; const float* b_ff1  = (const float*)d_in[31];
  const float* w_ff2  = (const float*)d_in[32]; const float* b_ff2  = (const float*)d_in[33];
  const float* fc_w   = (const float*)d_in[34]; const float* fc_b   = (const float*)d_in[35];

  const int lvlH[3] = {100, 50, 25}, lvlW[3] = {152, 76, 38};
  const int lvlC[3] = {512, 1024, 2048}, lvlS[3] = {0, 15200, 19000};
  const long NLq = (long)NB * LQ_TOT;

  char* base = (char*)d_ws;
  size_t cur = 0;
  auto allocB = [&](size_t bytes) {
    void* p = base + cur;
    cur += (bytes + 255) & ~(size_t)255;
    return p;
  };
  float*          P      = (float*)allocB((size_t)NLq * 256 * 4);
  float*          srcb   = (float*)allocB((size_t)NLq * 256 * 4);
  unsigned short* srcb16 = (unsigned short*)allocB((size_t)NLq * 256 * 2);
  unsigned short* qb16   = (unsigned short*)allocB((size_t)NLq * 256 * 2);
  float*          valb   = (float*)allocB((size_t)NLq * 256 * 4);
  float*          offb   = (float*)allocB((size_t)NLq * 192 * 4);
  float*          awlb   = (float*)allocB((size_t)NLq * 96 * 4);
  unsigned short* accb16 = (unsigned short*)allocB((size_t)NLq * 256 * 2);
  float*          attnb  = (float*)allocB((size_t)NLq * 256 * 4);
  float*          xb     = (float*)allocB((size_t)NLq * 256 * 4);
  unsigned short* xb16   = (unsigned short*)allocB((size_t)NLq * 256 * 2);
  unsigned short* ffb16  = (unsigned short*)allocB((size_t)NLq * 1024 * 2);
  float*          ff2b   = (float*)allocB((size_t)NLq * 256 * 4);
  float*          x2b    = (float*)allocB((size_t)NLq * 256 * 4);
  unsigned short* pfb16  = (unsigned short*)allocB((size_t)NB * 1000 * 256 * 2);
  float*          statb  = (float*)allocB((size_t)3 * NB * 32 * 2 * 4);

  // bf16 weight copies
  unsigned short* wproj16[3];
  for (int l = 0; l < 3; ++l)
    wproj16[l] = (unsigned short*)allocB((size_t)256 * lvlC[l] * 2);
  unsigned short* wval16  = (unsigned short*)allocB((size_t)256 * 256 * 2);
  unsigned short* woff16  = (unsigned short*)allocB((size_t)192 * 256 * 2);
  unsigned short* wattn16 = (unsigned short*)allocB((size_t)96 * 256 * 2);
  unsigned short* wout16  = (unsigned short*)allocB((size_t)256 * 256 * 2);
  unsigned short* wff1_16 = (unsigned short*)allocB((size_t)1024 * 256 * 2);
  unsigned short* wff2_16 = (unsigned short*)allocB((size_t)256 * 1024 * 2);
  unsigned short* wfc16   = (unsigned short*)allocB((size_t)12544 * 256 * 2);

  dim3 blk(256);
  auto cvt = [&](const float* in, unsigned short* out, long n) {
    long n4 = n >> 2;
    cvt_bf16_kernel<<<dim3((unsigned)((n4 + 255) / 256)), blk, 0, stream>>>(in, out, n4);
  };
  // 0) pre-convert all weights to bf16
  for (int l = 0; l < 3; ++l) cvt(proj_w[l], wproj16[l], (long)256 * lvlC[l]);
  cvt(w_val,  wval16,  256 * 256);
  cvt(w_off,  woff16,  192 * 256);
  cvt(w_attn, wattn16, 96 * 256);
  cvt(w_out,  wout16,  256 * 256);
  cvt(w_ff1,  wff1_16, 1024 * 256);
  cvt(w_ff2,  wff2_16, 256 * 1024);
  cvt(fc_w,   wfc16,   12544L * 256);

  auto gemm_t = [&](const unsigned short* A, const unsigned short* Bw, const float* bias,
                    void* C, int M, int Nn, int K, int relu, int obf16) {
    dim3 grid((Nn + BN - 1) / BN, (M + BM - 1) / BM, 1);
    gemm_wmma<1><<<grid, blk, 0, stream>>>(A, Bw, bias, C, M, Nn, K,
                                           (long)K, 1L, 0L, 0L, relu, obf16);
  };

  // 1) level projections (1x1 conv as GEMM; f32 A channel-major: a_k_stride = HW)
  for (int l = 0; l < 3; ++l) {
    int HW = lvlH[l] * lvlW[l];
    dim3 grid((256 + BN - 1) / BN, (HW + BM - 1) / BM, NB);
    gemm_wmma<0><<<grid, blk, 0, stream>>>(
        feat[l], wproj16[l], proj_b[l], P + (size_t)lvlS[l] * 256,
        HW, 256, lvlC[l], 1L, (long)HW,
        (long)lvlC[l] * HW, (long)LQ_TOT * 256, 0, 0);
  }
  // 2) GroupNorm + positional encodings -> src(f32+bf16), q(bf16)
  for (int l = 0; l < 3; ++l) {
    int HW = lvlH[l] * lvlW[l];
    float* st = statb + (size_t)l * NB * 32 * 2;
    gn_stats_kernel<<<dim3(32, NB), blk, 0, stream>>>(P, st, HW, lvlS[l]);
    long tot = (long)HW * 256;
    gn_apply_pos_kernel<<<dim3((unsigned)((tot + 255) / 256), NB), blk, 0, stream>>>(
        P, st, gn_g[l], gn_b[l], level_emb + l * 256, srcb, srcb16, qb16,
        HW, lvlS[l], lvlH[l], lvlW[l]);
  }
  // 3) value / offset / attention-logit projections
  gemm_t(srcb16, wval16,  b_val,  valb, (int)NLq, 256, 256, 0, 0);
  gemm_t(qb16,   woff16,  b_off,  offb, (int)NLq, 192, 256, 0, 0);
  gemm_t(qb16,   wattn16, b_attn, awlb, (int)NLq,  96, 256, 0, 0);
  // 4) deformable multi-scale sampling (softmax fused) -> bf16 accumulator
  deform_sample_kernel<<<dim3((NB * LQ_TOT * 8 + 7) / 8), blk, 0, stream>>>(
      valb, offb, awlb, accb16);
  // 5) output projection + residual LayerNorm
  gemm_t(accb16, wout16, b_out, attnb, (int)NLq, 256, 256, 0, 0);
  add_ln_kernel<<<dim3((unsigned)NLq), blk, 0, stream>>>(srcb, attnb, ln1_g, ln1_b, xb, xb16);
  // 6) FFN (256 -> 1024 relu -> 256, intermediate kept bf16) + residual LayerNorm
  gemm_t(xb16,  wff1_16, b_ff1, ffb16, (int)NLq, 1024,  256, 1, 1);
  gemm_t(ffb16, wff2_16, b_ff2, ff2b,  (int)NLq,  256, 1024, 0, 0);
  add_ln_kernel<<<dim3((unsigned)NLq), blk, 0, stream>>>(xb, ff2b, ln2_g, ln2_b, x2b, (unsigned short*)nullptr);
  // 7) point sampling + final FC (relu) straight into d_out
  {
    long tot = (long)NB * 1000 * 256;
    point_sample_kernel<<<dim3((unsigned)((tot + 255) / 256)), blk, 0, stream>>>(
        x2b, pcoords, pfb16);
  }
  gemm_t(pfb16, wfc16, fc_b, (float*)d_out, NB * 1000, 12544, 256, 1, 0);
  // 8) gt_classes passthrough (second tuple element, raw 4-byte copy)
  hipMemcpyAsync((char*)d_out + (size_t)NB * 1000 * 12544 * sizeof(float), gtcls,
                 (size_t)NB * 1000 * sizeof(int), hipMemcpyDeviceToDevice, stream);

  (void)in_sizes; (void)n_in; (void)out_size; (void)ws_size;
}
